// Attention_53927609369165
// MI455X (gfx1250) — compile-verified, matrix operand order checked by reading, not run on hardware
//
#include <hip/hip_runtime.h>
#include <math.h>

#define DIM 128
#define B_SZ 32
#define L_SZ 8192
#define NCHUNKS 64
#define CHUNK_ROWS 128
// 128 + 8 bf16 pad -> 272B row stride; 272/4 = 68 banks -> 4-bank rotation per row,
// so 16 lanes doing 16B ds_load_b128 from consecutive rows hit disjoint banks.
#define KT_STRIDE 136

typedef __attribute__((ext_vector_type(16))) __bf16 v16bf;
typedef __attribute__((ext_vector_type(8)))  float  v8f;
typedef unsigned int u32x4 __attribute__((ext_vector_type(4)));   // 16-byte aligned -> ds_load_b128

union Frag { u32x4 q[2]; v16bf v; };

__device__ __forceinline__ unsigned short f2bf(float f) {
  unsigned int u = __float_as_uint(f);
  u += 0x7FFFu + ((u >> 16) & 1u);   // round-to-nearest-even f32 -> bf16
  return (unsigned short)(u >> 16);
}

// CDNA5 has V_TANH_F32 (TRANS op). Use it; fall back to a branchless exp2 form.
__device__ __forceinline__ float fast_tanh(float x) {
#if __has_builtin(__builtin_amdgcn_tanhf)
  return __builtin_amdgcn_tanhf(x);
#elif __has_builtin(__builtin_amdgcn_tanh_f32)
  return __builtin_amdgcn_tanh_f32(x);
#else
  // tanh(x) = 1 - 2/(exp2(2*log2e*x)+1); exact limits at +/-inf, branch-free
  const float e = __builtin_amdgcn_exp2f(2.8853900818f * x);
  return 1.f - 2.f * __builtin_amdgcn_rcpf(e + 1.f);
#endif
}

__device__ __forceinline__ float fast_exp(float x) {
#if __has_builtin(__builtin_amdgcn_exp2f)
  return __builtin_amdgcn_exp2f(1.44269504089f * x);   // v_exp_f32, branchless
#else
  return __expf(x);
#endif
}

// ---- kernel 1: qW[b,h] = sum_d q[b,d] * Wq[d,h]  (tiny) ----
__global__ __launch_bounds__(128) void attn_qw(
    const float* __restrict__ q, const float* __restrict__ W, float* __restrict__ qw)
{
  __shared__ float qs[DIM];
  const int b = blockIdx.x, h = threadIdx.x;
  qs[h] = q[b * DIM + h];
  __syncthreads();
  float s = 0.f;
  for (int d0 = 0; d0 < DIM; ++d0) s += qs[d0] * W[d0 * DIM + h];
  qw[b * DIM + h] = s;
}

// ---- kernel 2: per-(b,chunk) scores via WMMA + chunk softmax + weighted v ----
__global__ __launch_bounds__(256) void attn_main(
    const float* __restrict__ k, const float* __restrict__ v,
    const float* __restrict__ W, const float* __restrict__ qw,
    const float* __restrict__ vparam,
    float* __restrict__ pm, float* __restrict__ ps, float* __restrict__ pacc)
{
  extern __shared__ unsigned char smem_raw[];
  unsigned short* kt  = (unsigned short*)smem_raw;        // [128][136] bf16 k tile
  unsigned short* wkT = kt + CHUNK_ROWS * KT_STRIDE;      // [128][136] bf16 Wk^T (h-major)
  float*  qws    = (float*)(wkT + DIM * KT_STRIDE);       // 128
  float*  vps    = qws + DIM;                             // 128
  float*  scores = vps + DIM;                             // 128
  float*  wts    = scores + CHUNK_ROWS;                   // 128
  float*  red    = wts + CHUNK_ROWS;                      // 128
  float4* accbuf = (float4*)(red + CHUNK_ROWS);           // 256 float4 (4096 B, 16B-aligned)

  const int b     = blockIdx.y;
  const int chunk = blockIdx.x;
  const int tid   = threadIdx.x;
  const int r0    = chunk * CHUNK_ROWS;

  // Wk^T -> LDS as bf16 (W rows 128..255; coalesced read over h)
  for (int i = tid; i < DIM * DIM; i += 256) {
    const int d = i >> 7, h = i & (DIM - 1);
    wkT[h * KT_STRIDE + d] = f2bf(W[(DIM + d) * DIM + h]);
  }
  if (tid < DIM) { qws[tid] = qw[b * DIM + tid]; vps[tid] = vparam[tid]; }

  // k tile -> LDS as bf16 (coalesced float4 reads, packed b64 LDS stores)
  const float* kb = k + ((size_t)b * L_SZ + r0) * DIM;
  for (int i = tid; i < CHUNK_ROWS * (DIM / 4); i += 256) {
    const int r = i >> 5, c4 = i & 31;
    const float4 x = ((const float4*)(kb + (size_t)r * DIM))[c4];
    const unsigned long long pkt =
        (unsigned long long)f2bf(x.x)        |
        ((unsigned long long)f2bf(x.y) << 16) |
        ((unsigned long long)f2bf(x.z) << 32) |
        ((unsigned long long)f2bf(x.w) << 48);
    *(unsigned long long*)&kt[r * KT_STRIDE + c4 * 4] = pkt;
  }
  __syncthreads();

  // Each wave owns a 16-row band.  A frag (ISA 16-bit 16x32 layout):
  // lane m=lane%16, g=lane/16: halfs 0..7 -> K = ks*32 + g*8 + {0..7},
  //                            halfs 8..15 -> K = ks*32 + 16 + g*8 + {0..7}.
  const int wave = tid >> 5, lane = tid & 31;
  const int n = lane & 15, g = lane >> 4;
  const int arow = wave * 16 + n;

  Frag fa[4];
  #pragma unroll
  for (int ks = 0; ks < 4; ++ks) {
    const int c0 = ks * 32 + g * 8;
    fa[ks].q[0] = *(const u32x4*)&kt[arow * KT_STRIDE + c0];
    fa[ks].q[1] = *(const u32x4*)&kt[arow * KT_STRIDE + c0 + 16];
  }

  float sp[8];
  #pragma unroll
  for (int j = 0; j < 8; ++j) sp[j] = 0.f;

  #pragma unroll
  for (int ht = 0; ht < 8; ++ht) {
    v8f c = {0.f, 0.f, 0.f, 0.f, 0.f, 0.f, 0.f, 0.f};
    const int hrow = ht * 16 + n;          // B frag: lane n = N column, read Wk^T row h
    #pragma unroll
    for (int ks = 0; ks < 4; ++ks) {
      const int c0 = ks * 32 + g * 8;
      Frag fb;
      fb.q[0] = *(const u32x4*)&wkT[hrow * KT_STRIDE + c0];
      fb.q[1] = *(const u32x4*)&wkT[hrow * KT_STRIDE + c0 + 16];
      c = __builtin_amdgcn_wmma_f32_16x16x32_bf16(false, fa[ks].v, false, fb.v,
                                                  (short)0, c, false, false);
    }
    // C layout: VGPR j, lanes 0-15 -> M=j, lanes 16-31 -> M=j+8; N = lane%16.
    const float qn = qws[hrow];
    const float vn = vps[hrow];
    #pragma unroll
    for (int j = 0; j < 8; ++j) sp[j] += fast_tanh(c[j] + qn) * vn;
  }

  // reduce over N (16-lane butterfly within each half-wave)
  #pragma unroll
  for (int j = 0; j < 8; ++j) {
    float x = sp[j];
    x += __shfl_xor(x, 1, 16);
    x += __shfl_xor(x, 2, 16);
    x += __shfl_xor(x, 4, 16);
    x += __shfl_xor(x, 8, 16);
    sp[j] = x;
  }
  if (n == 0) {
    #pragma unroll
    for (int j = 0; j < 8; ++j) scores[wave * 16 + g * 8 + j] = sp[j];
  }
  __syncthreads();

  // chunk-local softmax stats (max, sum of exp)
  if (tid < CHUNK_ROWS) red[tid] = scores[tid];
  __syncthreads();
  for (int s = CHUNK_ROWS / 2; s > 0; s >>= 1) {
    if (tid < s) red[tid] = fmaxf(red[tid], red[tid + s]);
    __syncthreads();
  }
  const float m = red[0];
  __syncthreads();
  if (tid < CHUNK_ROWS) { const float w0 = fast_exp(scores[tid] - m); wts[tid] = w0; red[tid] = w0; }
  __syncthreads();
  for (int s = CHUNK_ROWS / 2; s > 0; s >>= 1) {
    if (tid < s) red[tid] += red[tid + s];
    __syncthreads();
  }
  const float ssum = red[0];

  // weighted accumulation of v rows: 32 threads * float4 cover one row (b128 loads),
  // 8 row-groups in parallel, reduced through LDS.
  const float* vb = v + ((size_t)b * L_SZ + r0) * DIM;
  const int c4 = tid & 31;    // float4 column
  const int rg = tid >> 5;    // row group 0..7
  float4 a4 = make_float4(0.f, 0.f, 0.f, 0.f);
  for (int r = rg; r < CHUNK_ROWS; r += 8) {
    const float w0 = wts[r];
    const float4 x = ((const float4*)(vb + (size_t)r * DIM))[c4];
    a4.x += w0 * x.x; a4.y += w0 * x.y; a4.z += w0 * x.z; a4.w += w0 * x.w;
  }
  accbuf[tid] = a4;
  __syncthreads();
  if (tid < 32) {
    float4 t = accbuf[c4];
    #pragma unroll
    for (int gg = 1; gg < 8; ++gg) {
      const float4 x = accbuf[gg * 32 + c4];
      t.x += x.x; t.y += x.y; t.z += x.z; t.w += x.w;
    }
    const int idx = b * NCHUNKS + chunk;
    ((float4*)(pacc + (size_t)idx * DIM))[c4] = t;
    if (tid == 0) { pm[idx] = m; ps[idx] = ssum; }
  }
}

// ---- kernel 3: merge chunk partials with max-rescaling ----
__global__ __launch_bounds__(128) void attn_reduce(
    const float* __restrict__ pm, const float* __restrict__ ps,
    const float* __restrict__ pacc, float* __restrict__ out)
{
  const int b = blockIdx.x, d = threadIdx.x;
  float M = -3.402823466e38f;
  for (int c = 0; c < NCHUNKS; ++c) M = fmaxf(M, pm[b * NCHUNKS + c]);
  float S = 0.f, a = 0.f;
  for (int c = 0; c < NCHUNKS; ++c) {
    const float e = fast_exp(pm[b * NCHUNKS + c] - M);
    S += ps[b * NCHUNKS + c] * e;
    a += pacc[(size_t)(b * NCHUNKS + c) * DIM + d] * e;
  }
  out[b * DIM + d] = a / S;   // softmax + reference's redundant renorm (identity)
}

extern "C" void kernel_launch(void* const* d_in, const int* in_sizes, int n_in,
                              void* d_out, int out_size, void* d_ws, size_t ws_size,
                              hipStream_t stream) {
  const float* q  = (const float*)d_in[0];
  const float* k  = (const float*)d_in[1];
  const float* v  = (const float*)d_in[2];
  const float* W  = (const float*)d_in[3];
  const float* vp = (const float*)d_in[4];
  float* out = (float*)d_out;

  // workspace layout (floats): qw[32*128] | pm[32*64] | ps[32*64] | pacc[32*64*128]
  // total = 270336 floats = 1,081,344 bytes
  float* wsf  = (float*)d_ws;
  float* qw   = wsf;
  float* pm   = wsf + B_SZ * DIM;                       // 4096
  float* ps   = pm  + B_SZ * NCHUNKS;                   // +2048
  float* pacc = wsf + B_SZ * DIM + 2 * B_SZ * NCHUNKS;  // 8192

  const int SMEM_BYTES = 2 * (CHUNK_ROWS * KT_STRIDE + DIM * KT_STRIDE)  // bf16 tiles: 69632
                       + 5 * CHUNK_ROWS * 4                              // qws/vps/scores/wts/red
                       + 256 * 16;                                       // accbuf float4 -> 76288 B
  (void)hipFuncSetAttribute(reinterpret_cast<const void*>(&attn_main),
                            hipFuncAttributeMaxDynamicSharedMemorySize, SMEM_BYTES);

  attn_qw   <<<dim3(B_SZ),          dim3(DIM), 0,          stream>>>(q, W, qw);
  attn_main <<<dim3(NCHUNKS, B_SZ), dim3(256), SMEM_BYTES, stream>>>(k, v, W, qw, vp, pm, ps, pacc);
  attn_reduce<<<dim3(B_SZ),         dim3(DIM), 0,          stream>>>(pm, ps, pacc, out);
}
